// AttentionTail_13005160973212
// MI455X (gfx1250) — matrix operand order repaired and checked by reading, not compiled, not run on hardware
//
#include <hip/hip_runtime.h>
#include <hip/hip_bf16.h>

typedef __attribute__((ext_vector_type(16))) _Float16 v16h;
typedef __attribute__((ext_vector_type(2)))  _Float16 h2;
typedef __attribute__((ext_vector_type(8)))  float    v8f;

#define WMMA_F16(a, b, c) \
  __builtin_amdgcn_wmma_f32_16x16x32_f16(false, (a), false, (b), (short)0, (c), false, false)

// ---------------------------------------------------------------------------
// Tile loaders following the CDNA5 ISA 16-bit WMMA VGPR layouts (CDNA5 ISA):
//  A (16x32 MxK): lane = (M = lane&15, half = lane>>4);
//                 halves[0..7]  = row[k0 + 8*half .. +7]
//                 halves[8..15] = row[k0 + 16 + 8*half .. +7]
//  B (32x16 KxN): lane = (N = lane&15, half = lane>>4);
//                 halves[0..7] = col[k0 + 8*half .. +7], halves[8..15] = col[k0 + 16 + 8*half .. +7]
//  C/D (16x16):   VGPR j: lanes 0-15 -> M=j, lanes 16-31 -> M=8+j; N = lane&15
// ---------------------------------------------------------------------------

__device__ __forceinline__ v16h load_a_f32(const float* __restrict__ rowPtr, int half, bool valid) {
  v16h r;
  if (valid) {
    const float* p0 = rowPtr + half * 8;
    const float* p1 = rowPtr + 16 + half * 8;
#pragma unroll
    for (int i = 0; i < 8; ++i) { r[i] = (_Float16)p0[i]; r[8 + i] = (_Float16)p1[i]; }
  } else {
#pragma unroll
    for (int i = 0; i < 16; ++i) r[i] = (_Float16)0.0f;
  }
  return r;
}

__device__ __forceinline__ v16h load_a_f16(const _Float16* __restrict__ rowPtr, int half, bool valid) {
  v16h r;
  if (valid) {
#pragma unroll
    for (int i = 0; i < 8; ++i) { r[i] = rowPtr[half * 8 + i]; r[8 + i] = rowPtr[16 + half * 8 + i]; }
  } else {
#pragma unroll
    for (int i = 0; i < 16; ++i) r[i] = (_Float16)0.0f;
  }
  return r;
}

__device__ __forceinline__ v16h load_b_f32(const float* __restrict__ colPtr, int half) {
  v16h r;
  const float* p = colPtr + half * 8;
#pragma unroll
  for (int i = 0; i < 8; ++i) { r[i] = (_Float16)p[i]; r[8 + i] = (_Float16)p[16 + i]; }
  return r;
}

__device__ __forceinline__ v16h load_b_f16(const _Float16* __restrict__ colPtr, int half) {
  v16h r;
  const _Float16* p = colPtr + half * 8;
#pragma unroll
  for (int i = 0; i < 8; ++i) { r[i] = p[i]; r[8 + i] = p[16 + i]; }
  return r;
}

// ---------------------------------------------------------------------------
// Kernel 1: Y[M x 256](f16) = X[M x 256](f32) @ W[256 x 256](f32).T
// One wave computes a 16(M) x 64(N) tile: 4 accumulators, 8 K-steps of 32.
// ---------------------------------------------------------------------------
__global__ __launch_bounds__(128) void proj_gemm_f16out(
    const float* __restrict__ X, const float* __restrict__ W,
    _Float16* __restrict__ Y, int M) {
  const int K = 256, N = 256;
  const int lane = threadIdx.x & 31;
  const int wave = blockIdx.x * 4 + (threadIdx.x >> 5);
  const int nTiles = N / 64;                 // 4
  const int mTiles = (M + 15) >> 4;
  if (wave >= mTiles * nTiles) return;
  const int m0 = (wave / nTiles) * 16;
  const int n0 = (wave % nTiles) * 64;
  const int half = lane >> 4;
  const int mrow = m0 + (lane & 15);
  const int ncol = lane & 15;
  const bool aValid = mrow < M;
  const float* aRow = X + (size_t)mrow * K;

  v8f acc0 = {}, acc1 = {}, acc2 = {}, acc3 = {};
#pragma unroll
  for (int kk = 0; kk < K; kk += 32) {
    v16h a = load_a_f32(aRow + kk, half, aValid);
    v16h b0 = load_b_f32(W + (size_t)(n0 +  0 + ncol) * K + kk, half);
    v16h b1 = load_b_f32(W + (size_t)(n0 + 16 + ncol) * K + kk, half);
    v16h b2 = load_b_f32(W + (size_t)(n0 + 32 + ncol) * K + kk, half);
    v16h b3 = load_b_f32(W + (size_t)(n0 + 48 + ncol) * K + kk, half);
    acc0 = WMMA_F16(a, b0, acc0);
    acc1 = WMMA_F16(a, b1, acc1);
    acc2 = WMMA_F16(a, b2, acc2);
    acc3 = WMMA_F16(a, b3, acc3);
  }
  v8f* accs[4] = {&acc0, &acc1, &acc2, &acc3};
#pragma unroll
  for (int t = 0; t < 4; ++t) {
#pragma unroll
    for (int j = 0; j < 8; ++j) {
      int r = m0 + half * 8 + j;
      if (r < M) Y[(size_t)r * N + (n0 + t * 16 + ncol)] = (_Float16)((*accs[t])[j]);
    }
  }
}

// ---------------------------------------------------------------------------
// Kernel 2: attention + per-level head-mix (Wl, bl) + ReLU, fused.
//   attn[b,n,l,h] = scale * sum_d Q[b,n,h*128+d] * Kp[b,l,h*128+d]
//   sl[h'] = relu(sum_h attn[h]*Wl[lvl][h'][h] + bl[lvl][h'])
// Stored interleaved f16: sl[((b*300+n)*L + l)*2 + h'] as packed h2 (4B).
// One wave: 16(n) x 64(l) tile for BOTH heads -> 8 f32 acc tiles, 32 WMMAs.
// All level boundaries (0,16384,20480,21504,21760) are multiples of 64, so a
// 64-wide l-tile never straddles levels.
// ---------------------------------------------------------------------------
__global__ __launch_bounds__(128) void attn_mix_kernel(
    const _Float16* __restrict__ Q,   // (B*300) x 256, heads in col halves
    const _Float16* __restrict__ Kp,  // (B*L)   x 256
    const float* __restrict__ Wl,     // 4 x 2 x 2
    const float* __restrict__ bl,     // 4 x 2
    _Float16* __restrict__ sl) {
  const int Nq = 300, L = 21760, C = 256;
  const int mTiles = 19, lTiles = 340;                // ceil(300/16), 21760/64
  const int lane = threadIdx.x & 31;
  int wave = blockIdx.x * 4 + (threadIdx.x >> 5);
  if (wave >= 2 * mTiles * lTiles) return;
  const int b = wave / (mTiles * lTiles);
  int r = wave % (mTiles * lTiles);
  const int m0 = (r / lTiles) * 16;
  const int l0 = (r % lTiles) * 64;
  const int half = lane >> 4;
  const int ncol = lane & 15;
  const int mrow = m0 + (lane & 15);
  const bool aValid = mrow < Nq;
  const _Float16* qRow = Q + (size_t)(b * Nq + mrow) * C;

  v8f acc[2][4];
#pragma unroll
  for (int h = 0; h < 2; ++h)
#pragma unroll
    for (int t = 0; t < 4; ++t) acc[h][t] = {};

#pragma unroll
  for (int kk = 0; kk < 128; kk += 32) {
    v16h a0 = load_a_f16(qRow + kk, half, aValid);         // head 0
    v16h a1 = load_a_f16(qRow + 128 + kk, half, aValid);   // head 1
#pragma unroll
    for (int t = 0; t < 4; ++t) {
      const _Float16* kRow = Kp + (size_t)(b * L + l0 + t * 16 + ncol) * C;
      v16h b0 = load_b_f16(kRow + kk, half);
      v16h b1 = load_b_f16(kRow + 128 + kk, half);
      acc[0][t] = WMMA_F16(a0, b0, acc[0][t]);
      acc[1][t] = WMMA_F16(a1, b1, acc[1][t]);
    }
  }

  // level of this l-tile (uniform across the wave)
  int lvl = (l0 < 16384) ? 0 : (l0 < 20480) ? 1 : (l0 < 21504) ? 2 : 3;
  const float w00 = Wl[lvl * 4 + 0], w01 = Wl[lvl * 4 + 1];
  const float w10 = Wl[lvl * 4 + 2], w11 = Wl[lvl * 4 + 3];
  const float bl0 = bl[lvl * 2 + 0], bl1 = bl[lvl * 2 + 1];
  const float scale = 0.088388347648318447f;   // 128^-0.5

#pragma unroll
  for (int t = 0; t < 4; ++t) {
    int l = l0 + t * 16 + ncol;
#pragma unroll
    for (int j = 0; j < 8; ++j) {
      int row = m0 + half * 8 + j;
      if (row < Nq) {
        float a0v = acc[0][t][j] * scale;
        float a1v = acc[1][t][j] * scale;
        h2 s;
        s[0] = (_Float16)fmaxf(w00 * a0v + w01 * a1v + bl0, 0.0f);
        s[1] = (_Float16)fmaxf(w10 * a0v + w11 * a1v + bl1, 0.0f);
        *reinterpret_cast<h2*>(sl + ((size_t)(b * Nq + row) * L + l) * 2) = s;
      }
    }
  }
}

// ---------------------------------------------------------------------------
// Kernel 3: bilinear upsample of all 4 levels + Wf dot + bf + ReLU, fused.
// Half-pixel centers with edge clamp (matches jax.image.resize 'bilinear'
// for upsampling). Level 0 (128->128) is an exact identity with half-pixel
// centers -> single tap. One thread per output pixel; the per-(b,n) 87KB sl
// slice is L2-resident across its 16384 output pixels.
// ---------------------------------------------------------------------------
__global__ __launch_bounds__(256) void tail_kernel(
    const _Float16* __restrict__ sl, const float* __restrict__ Wf,
    const float* __restrict__ bf, float* __restrict__ out) {
  const int L = 21760;
  const size_t total = (size_t)600 * 16384;
  size_t idx = (size_t)blockIdx.x * 256 + threadIdx.x;
  if (idx >= total) return;
  const int p = (int)(idx & 16383);
  const size_t bn = idx >> 14;            // b*300 + n
  const int y = p >> 7, x = p & 127;
  const h2* base = (const h2*)(sl) + bn * (size_t)L;

  float acc = bf[0];

  // level 0: 128 -> 128 identity
  {
    h2 v = base[p];
    acc += Wf[0] * (float)v[0] + Wf[1] * (float)v[1];
  }

  int off = 16384;
#pragma unroll
  for (int lvl = 1; lvl < 4; ++lvl) {
    const int sz = 128 >> lvl;
    const float inv = 1.0f / (float)(1 << lvl);
    float sy = (y + 0.5f) * inv - 0.5f;
    float sx = (x + 0.5f) * inv - 0.5f;
    float fy0 = floorf(sy), fx0 = floorf(sx);
    float wy = sy - fy0, wx = sx - fx0;
    int y0 = min(max((int)fy0, 0), sz - 1);
    int y1 = min(max((int)fy0 + 1, 0), sz - 1);
    int x0 = min(max((int)fx0, 0), sz - 1);
    int x1 = min(max((int)fx0 + 1, 0), sz - 1);
    const h2* r0 = base + off + (size_t)y0 * sz;
    const h2* r1 = base + off + (size_t)y1 * sz;
    h2 v00 = r0[x0], v01 = r0[x1], v10 = r1[x0], v11 = r1[x1];
    float c00 = (1.0f - wy) * (1.0f - wx), c01 = (1.0f - wy) * wx;
    float c10 = wy * (1.0f - wx),          c11 = wy * wx;
    float f0 = c00 * (float)v00[0] + c01 * (float)v01[0]
             + c10 * (float)v10[0] + c11 * (float)v11[0];
    float f1 = c00 * (float)v00[1] + c01 * (float)v01[1]
             + c10 * (float)v10[1] + c11 * (float)v11[1];
    acc += Wf[lvl * 2 + 0] * f0 + Wf[lvl * 2 + 1] * f1;
    off += sz * sz;
  }
  out[idx] = fmaxf(acc, 0.0f);
}

// ---------------------------------------------------------------------------
extern "C" void kernel_launch(void* const* d_in, const int* in_sizes, int n_in,
                              void* d_out, int out_size, void* d_ws, size_t ws_size,
                              hipStream_t stream) {
  (void)in_sizes; (void)n_in; (void)out_size; (void)ws_size;
  const float* query = (const float*)d_in[0];   // (2,300,256)
  const float* key   = (const float*)d_in[1];   // (2,21760,256)
  const float* Wq    = (const float*)d_in[2];   // (256,256)
  const float* Wk    = (const float*)d_in[3];   // (256,256)
  const float* Wl    = (const float*)d_in[4];   // (4,2,2)
  const float* bl    = (const float*)d_in[5];   // (4,2)
  const float* Wf    = (const float*)d_in[6];   // (1,8)
  const float* bf    = (const float*)d_in[7];   // (1,)
  float* out = (float*)d_out;                   // (2,300,16384,1)

  const int Mq = 2 * 300;        // 600
  const int Mk = 2 * 21760;      // 43520

  // Workspace layout (all 256B aligned):
  char* ws = (char*)d_ws;
  _Float16* qproj = (_Float16*)ws;                         // 600*256   f16 =   307,200 B
  _Float16* kproj = (_Float16*)(ws + 307200);              // 43520*256 f16 = 22,282,240 B
  _Float16* slbuf = (_Float16*)(ws + 307200 + 22282240);   // 600*21760*2 f16 = 52,224,000 B

  // q / k projections (each wave = 16x64 tile, 4 waves per block)
  {
    int waves = ((Mq + 15) / 16) * 4;   // 152
    proj_gemm_f16out<<<(waves + 3) / 4, 128, 0, stream>>>(query, Wq, qproj, Mq);
  }
  {
    int waves = (Mk / 16) * 4;          // 10880
    proj_gemm_f16out<<<waves / 4, 128, 0, stream>>>(key, Wk, kproj, Mk);
  }

  // fused attention + level head-mix + relu
  {
    int waves = 2 * 19 * 340;           // 12920
    attn_mix_kernel<<<waves / 4, 128, 0, stream>>>(qproj, kproj, Wl, bl, slbuf);
  }

  // fused bilinear + Wf + bf + relu
  {
    size_t total = (size_t)600 * 16384; // 9,830,400
    tail_kernel<<<(unsigned)((total + 255) / 256), 256, 0, stream>>>(slbuf, Wf, bf, out);
  }
}
